// Seq2SeqBahdanau_35527969472789
// MI455X (gfx1250) — matrix operand order repaired and checked by reading, not compiled, not run on hardware
//
#include <hip/hip_runtime.h>
#include <cstdint>
#include <cstddef>

typedef unsigned short u16;
typedef __attribute__((ext_vector_type(16))) __bf16 bf16x16;
typedef __attribute__((ext_vector_type(8)))  float   f32x8;

union FragAB { uint4 q[2]; bf16x16 v; };

__device__ __forceinline__ f32x8 wmma_bf16(const FragAB& a, const FragAB& b, f32x8 c) {
  return __builtin_amdgcn_wmma_f32_16x16x32_bf16(false, a.v, false, b.v, (short)0, c, false, false);
}

__device__ __forceinline__ u16 f2bf(float x) {
  unsigned u = __float_as_uint(x);
  unsigned r = (u + 0x7FFFu + ((u >> 16) & 1u)) >> 16;
  return (u16)r;
}
__device__ __forceinline__ float bf2f(u16 h) { return __uint_as_float(((unsigned)h) << 16); }
__device__ __forceinline__ float sigf(float x) { return 1.0f / (1.0f + __expf(-x)); }

// LDS byte offset of a generic pointer that points into shared memory.
__device__ __forceinline__ unsigned lds_off(const void* p) {
  return (unsigned)(unsigned long long)(const __attribute__((address_space(3))) char*)p;
}

// ---------------------------------------------------------------------------
// Generic bf16 WMMA GEMM:  C[M,N] = A[M,K] @ W[N,K]^T (+bias)
// block = 256 threads (8 waves), block tile 32x256 (wave: 1 Mtile x 4 Ntiles)
// B-fragments are preloaded per k-step so loads stay in flight across WMMAs.
// MODE 0: plain store   MODE 1: tanh(x+bias)   MODE 2: scatter to [B,T,V] out
// ---------------------------------------------------------------------------
template <int MODE>
__global__ __launch_bounds__(256)
void k_gemm(const u16* __restrict__ A, const u16* __restrict__ W,
            const float* __restrict__ bias, float* __restrict__ C,
            int M, int N, int K, int Tdim, int Vdim) {
  const int lane = threadIdx.x & 31;
  const int wave = threadIdx.x >> 5;
  const int half = lane >> 4;
  const int l16  = lane & 15;
  const int mtile = blockIdx.y * 2 + (wave & 1);
  const int ntb   = blockIdx.x * 16 + (wave >> 1) * 4;
  const int arow  = mtile * 16 + l16;
  const u16* Ap = A + (size_t)arow * K;

  f32x8 acc[4];
#pragma unroll
  for (int j = 0; j < 4; ++j)
#pragma unroll
    for (int r = 0; r < 8; ++r) acc[j][r] = 0.0f;

  for (int k0 = 0; k0 < K; k0 += 32) {
    FragAB fa, fb[4];
    fa.q[0] = *reinterpret_cast<const uint4*>(Ap + k0 + half * 16);
    fa.q[1] = *reinterpret_cast<const uint4*>(Ap + k0 + half * 16 + 8);
#pragma unroll
    for (int j = 0; j < 4; ++j) {
      const u16* Wp = W + (size_t)((ntb + j) * 16 + l16) * K + k0 + half * 16;
      fb[j].q[0] = *reinterpret_cast<const uint4*>(Wp);
      fb[j].q[1] = *reinterpret_cast<const uint4*>(Wp + 8);
      __builtin_prefetch(Wp + 64, 0, 0);   // next 128B of this weight row
    }
#pragma unroll
    for (int j = 0; j < 4; ++j) acc[j] = wmma_bf16(fa, fb[j], acc[j]);
  }
#pragma unroll
  for (int j = 0; j < 4; ++j) {
    const int col = (ntb + j) * 16 + l16;
    const float bb = bias ? bias[col] : 0.0f;
#pragma unroll
    for (int r = 0; r < 8; ++r) {
      const int row = mtile * 16 + half * 8 + r;
      float v = acc[j][r] + bb;
      if (MODE == 0) {
        C[(size_t)row * N + col] = v;
      } else if (MODE == 1) {
        C[(size_t)row * N + col] = tanhf(v);
      } else {
        const int t = row >> 5, b = row & 31;
        C[((size_t)b * Tdim + t + 1) * Vdim + col] = v;
      }
    }
  }
}

// ---------------------------------------------------------------------------
// Encoder LSTM scan: one persistent 1024-thread block per direction.
// grid.x: 0 = forward, 1 = backward. Per step: h@Whh^T via WMMA (gates in LDS),
// then gate nonlinearities + state update. gx = X@Wih^T + b precomputed and
// DMA-staged into LDS with global_load_async_to_lds_b128, overlapped with the
// WMMA phase. Dynamic LDS: gates 128K + gx stage 128K + c 32K + h 16K = 304KB.
// ---------------------------------------------------------------------------
__global__ __launch_bounds__(1024)
void k_enc_scan(const float* __restrict__ gx_f, const float* __restrict__ gx_b,
                const u16* __restrict__ whh_f, const u16* __restrict__ whh_b,
                u16* __restrict__ Y, u16* __restrict__ hlast, u16* __restrict__ clast) {
  const int dir = blockIdx.x;
  const float* gx = dir ? gx_b : gx_f;
  const u16* whh  = dir ? whh_b : whh_f;
  const int colofs = dir ? 256 : 0;

  extern __shared__ char smem[];
  float* gates = (float*)smem;                 // 32*1024 f32
  float* cst   = gates + 32 * 1024;            // 32*256  f32
  u16*   hbf   = (u16*)(cst + 32 * 256);       // 32*256  bf16
  float* gxbuf = (float*)(hbf + 32 * 256);     // 32*1024 f32 (async stage)

  const int tid = threadIdx.x;
  for (int i = tid; i < 8192; i += 1024) { cst[i] = 0.0f; hbf[i] = 0; }
  __syncthreads();

  const unsigned gx_lds = lds_off(gxbuf);
  const int lane = tid & 31, wave = tid >> 5;
  const int half = lane >> 4, l16 = lane & 15;
  const int mtile = wave & 1;
  const int ntb   = (wave >> 1) * 4;
  const int arow  = mtile * 16 + l16;

  for (int t = 0; t < 64; ++t) {
    const int s = dir ? (63 - t) : t;

    // --- async DMA: stage gx[s] (32x1024 f32 = 128KB) into LDS ---
    {
      const char* gsrc = (const char*)(gx + (size_t)s * 32768);
#pragma unroll
      for (int j = 0; j < 8; ++j) {
        const int c = (tid + j * 1024) * 16;   // 16B chunks
        asm volatile("global_load_async_to_lds_b128 %0, %1, off"
                     :: "v"(gx_lds + (unsigned)c), "v"(gsrc + c) : "memory");
      }
    }

    // --- recurrent GEMM: gates += h @ Whh^T (overlaps the DMA above) ---
    f32x8 acc[4];
#pragma unroll
    for (int j = 0; j < 4; ++j)
#pragma unroll
      for (int r = 0; r < 8; ++r) acc[j][r] = 0.0f;

#pragma unroll 2
    for (int k0 = 0; k0 < 256; k0 += 32) {
      FragAB fa, fb[4];
      fa.q[0] = *reinterpret_cast<const uint4*>(hbf + arow * 256 + k0 + half * 16);
      fa.q[1] = *reinterpret_cast<const uint4*>(hbf + arow * 256 + k0 + half * 16 + 8);
#pragma unroll
      for (int j = 0; j < 4; ++j) {
        const u16* Wp = whh + (size_t)((ntb + j) * 16 + l16) * 256 + k0 + half * 16;
        fb[j].q[0] = *reinterpret_cast<const uint4*>(Wp);
        fb[j].q[1] = *reinterpret_cast<const uint4*>(Wp + 8);
      }
#pragma unroll
      for (int j = 0; j < 4; ++j) acc[j] = wmma_bf16(fa, fb[j], acc[j]);
    }

    // wait for the gx DMA of this wave, then barrier so all slices are in LDS
    asm volatile("s_wait_asynccnt 0x0" ::: "memory");
    __syncthreads();

#pragma unroll
    for (int j = 0; j < 4; ++j) {
      const int col = (ntb + j) * 16 + l16;
#pragma unroll
      for (int r = 0; r < 8; ++r) {
        const int row = mtile * 16 + half * 8 + r;
        gates[row * 1024 + col] = acc[j][r] + gxbuf[row * 1024 + col];
      }
    }
    __syncthreads();

#pragma unroll
    for (int k = 0; k < 8; ++k) {
      const int cell = tid + k * 1024;
      const int b = cell >> 8, n = cell & 255;
      const float gi = gates[b * 1024 + n];
      const float gf = gates[b * 1024 + n + 256];
      const float gg = gates[b * 1024 + n + 512];
      const float go = gates[b * 1024 + n + 768];
      const float cn = sigf(gf) * cst[cell] + sigf(gi) * tanhf(gg);
      const float hn = sigf(go) * tanhf(cn);
      cst[cell] = cn;
      const u16 hb = f2bf(hn);
      hbf[cell] = hb;
      Y[((size_t)s * 32 + b) * 512 + colofs + n] = hb;
      if (t == 63) {
        hlast[b * 512 + colofs + n] = hb;
        clast[b * 512 + colofs + n] = f2bf(cn);
      }
    }
    __syncthreads();   // protects hbf (next GEMM) and gxbuf (next DMA)
  }
}

// ---------------------------------------------------------------------------
// Device-scope global barrier for the persistent decoder kernel.
// ---------------------------------------------------------------------------
__device__ __forceinline__ void gbar(unsigned* bar, int nblk) {
  __threadfence();
  __syncthreads();
  if (threadIdx.x == 0) {
    const unsigned gen = __hip_atomic_load(&bar[1], __ATOMIC_ACQUIRE, __HIP_MEMORY_SCOPE_AGENT);
    const unsigned arr = __hip_atomic_fetch_add(&bar[0], 1u, __ATOMIC_ACQ_REL, __HIP_MEMORY_SCOPE_AGENT);
    if (arr == (unsigned)(nblk - 1)) {
      __hip_atomic_store(&bar[0], 0u, __ATOMIC_RELAXED, __HIP_MEMORY_SCOPE_AGENT);
      __hip_atomic_fetch_add(&bar[1], 1u, __ATOMIC_ACQ_REL, __HIP_MEMORY_SCOPE_AGENT);
    } else {
      while (__hip_atomic_load(&bar[1], __ATOMIC_ACQUIRE, __HIP_MEMORY_SCOPE_AGENT) == gen) {
        __builtin_amdgcn_s_sleep(2);
      }
    }
  }
  __syncthreads();
}

// Decoder cell GEMM: gates[32,1024] = X[32,K] @ W[1024,K]^T + bias.
// 64 global waves (8 blocks x 8 waves), 2 tiles each; B preloaded per k-step.
__device__ __forceinline__ void dec_gemm_phase(const u16* __restrict__ X, int K,
                                               const u16* __restrict__ W,
                                               const float* __restrict__ bias,
                                               float* __restrict__ gates, int gtid) {
  const int lane = gtid & 31;
  const int wv   = gtid >> 5;   // 0..63
  const int half = lane >> 4, l16 = lane & 15;
  const int mtile = wv & 1;
  const int ntb   = (wv >> 1) * 2;
  const u16* Ap = X + (size_t)(mtile * 16 + l16) * K;
  f32x8 acc0, acc1;
#pragma unroll
  for (int r = 0; r < 8; ++r) { acc0[r] = 0.0f; acc1[r] = 0.0f; }
  for (int k0 = 0; k0 < K; k0 += 32) {
    FragAB fa, fb0, fb1;
    fa.q[0] = *reinterpret_cast<const uint4*>(Ap + k0 + half * 16);
    fa.q[1] = *reinterpret_cast<const uint4*>(Ap + k0 + half * 16 + 8);
    {
      const u16* Wp = W + (size_t)(ntb * 16 + l16) * K + k0 + half * 16;
      fb0.q[0] = *reinterpret_cast<const uint4*>(Wp);
      fb0.q[1] = *reinterpret_cast<const uint4*>(Wp + 8);
    }
    {
      const u16* Wp = W + (size_t)((ntb + 1) * 16 + l16) * K + k0 + half * 16;
      fb1.q[0] = *reinterpret_cast<const uint4*>(Wp);
      fb1.q[1] = *reinterpret_cast<const uint4*>(Wp + 8);
    }
    acc0 = wmma_bf16(fa, fb0, acc0);
    acc1 = wmma_bf16(fa, fb1, acc1);
  }
#pragma unroll
  for (int r = 0; r < 8; ++r) {
    const int row = mtile * 16 + half * 8 + r;
    const int c0 = ntb * 16 + l16, c1 = (ntb + 1) * 16 + l16;
    gates[row * 1024 + c0] = acc0[r] + bias[c0];
    gates[row * 1024 + c1] = acc1[r] + bias[c1];
  }
}

__device__ __forceinline__ void dec_update(int l, int t, int gtid,
                                           const float* __restrict__ gates,
                                           float* __restrict__ hst, float* __restrict__ cstv,
                                           u16* __restrict__ x0, u16* __restrict__ x1,
                                           u16* __restrict__ x2, u16* __restrict__ xctx) {
#pragma unroll
  for (int k = 0; k < 4; ++k) {
    const int cell = gtid + k * 2048;   // 8192 cells over 2048 threads
    const int b = cell >> 8, n = cell & 255;
    const float gi = gates[b * 1024 + n];
    const float gf = gates[b * 1024 + n + 256];
    const float gg = gates[b * 1024 + n + 512];
    const float go = gates[b * 1024 + n + 768];
    const float cn = sigf(gf) * cstv[l * 8192 + cell] + sigf(gi) * tanhf(gg);
    const float hn = sigf(go) * tanhf(cn);
    cstv[l * 8192 + cell] = cn;
    const u16 hb = f2bf(hn);
    if (l == 0) {
      x1[b * 512 + n] = hb;
      x0[b * 1024 + 768 + n] = hb;
    } else if (l == 1) {
      x2[b * 512 + n] = hb;
      x1[b * 512 + 256 + n] = hb;
    } else {
      hst[16384 + cell] = hn;                 // h_top fp32 for next attention
      x2[b * 512 + 256 + n] = hb;
      xctx[((size_t)t * 32 + b) * 768 + n] = hb;
    }
  }
}

// ---------------------------------------------------------------------------
// Persistent decoder: 8 blocks x 256 threads, 35 teacher-forced steps.
// Attention (VALU) -> L0/L1/L2 LSTM cells (WMMA GEMM + update), global barriers.
// ---------------------------------------------------------------------------
__global__ __launch_bounds__(256)
void k_decoder(const int* __restrict__ src, const u16* __restrict__ enc_out,
               const float* __restrict__ enc_proj, const float* __restrict__ attn_dec,
               const float* __restrict__ vw, const u16* __restrict__ embT,
               const u16* __restrict__ w0, const u16* __restrict__ w1, const u16* __restrict__ w2,
               const float* __restrict__ b0, const float* __restrict__ b1, const float* __restrict__ b2,
               float* __restrict__ hst, float* __restrict__ cstv,
               u16* __restrict__ x0, u16* __restrict__ x1, u16* __restrict__ x2,
               float* __restrict__ gates, u16* __restrict__ xctx, unsigned* __restrict__ bar) {
  __shared__ float sdp[256];
  __shared__ float ssc[64];
  const int tid = threadIdx.x, bid = blockIdx.x;
  const int gtid = bid * 256 + tid;

  // seed h-tails of the layer input buffers from the bridge state
  for (int cell = gtid; cell < 8192; cell += 2048) {
    const int b = cell >> 8, n = cell & 255;
    x0[b * 1024 + 768 + n] = f2bf(hst[cell]);
    x1[b * 512 + 256 + n]  = f2bf(hst[8192 + cell]);
    x2[b * 512 + 256 + n]  = f2bf(hst[16384 + cell]);
  }
  gbar(bar, 8);

  for (int t = 0; t < 35; ++t) {
    // ---- attention: each block handles 4 batch rows ----
    for (int nb = 0; nb < 4; ++nb) {
      const int b = bid * 4 + nb;
      {
        const float* hp = hst + 16384 + b * 256;
        const float* wp = attn_dec + tid * 256;
        float a = 0.0f;
        for (int k = 0; k < 256; ++k) a += hp[k] * wp[k];
        sdp[tid] = a;
      }
      __syncthreads();
      if (tid < 64) {
        const int s = tid;
        const float* ep = enc_proj + ((size_t)s * 32 + b) * 256;
        float a = 0.0f;
        for (int k = 0; k < 256; ++k) a += tanhf(ep[k] + sdp[k]) * vw[k];
        ssc[s] = (src[b * 64 + s] == 0) ? -1e9f : a;
      }
      __syncthreads();
      if (tid == 0) {
        float mx = -3.4e38f;
        for (int s2 = 0; s2 < 64; ++s2) mx = fmaxf(mx, ssc[s2]);
        float sum = 0.0f;
        for (int s2 = 0; s2 < 64; ++s2) { const float e = __expf(ssc[s2] - mx); ssc[s2] = e; sum += e; }
        const float inv = 1.0f / sum;
        for (int s2 = 0; s2 < 64; ++s2) ssc[s2] *= inv;
      }
      __syncthreads();
#pragma unroll
      for (int dd = 0; dd < 2; ++dd) {
        const int d = tid + dd * 256;
        float a = 0.0f;
        for (int s2 = 0; s2 < 64; ++s2) a += ssc[s2] * bf2f(enc_out[((size_t)s2 * 32 + b) * 512 + d]);
        const u16 cb = f2bf(a);
        x0[b * 1024 + 256 + d] = cb;
        xctx[((size_t)t * 32 + b) * 768 + 256 + d] = cb;
      }
      x0[b * 1024 + tid] = embT[((size_t)t * 32 + b) * 256 + tid];
      __syncthreads();
    }
    gbar(bar, 8);
    dec_gemm_phase(x0, 1024, w0, b0, gates, gtid); gbar(bar, 8);
    dec_update(0, t, gtid, gates, hst, cstv, x0, x1, x2, xctx); gbar(bar, 8);
    dec_gemm_phase(x1, 512, w1, b1, gates, gtid); gbar(bar, 8);
    dec_update(1, t, gtid, gates, hst, cstv, x0, x1, x2, xctx); gbar(bar, 8);
    dec_gemm_phase(x2, 512, w2, b2, gates, gtid); gbar(bar, 8);
    dec_update(2, t, gtid, gates, hst, cstv, x0, x1, x2, xctx); gbar(bar, 8);
  }
}

// ---------------------------------------------------------------------------
// Prep kernels
// ---------------------------------------------------------------------------
__global__ void k_cvt(const float* __restrict__ a, u16* __restrict__ d, long n) {
  for (long i = (long)blockIdx.x * 256 + threadIdx.x; i < n; i += (long)gridDim.x * 256)
    d[i] = f2bf(a[i]);
}
__global__ void k_cvt_concat(const float* __restrict__ a, int ca,
                             const float* __restrict__ b, int cb,
                             u16* __restrict__ d, int rows) {
  const long n = (long)rows * (ca + cb);
  for (long i = (long)blockIdx.x * 256 + threadIdx.x; i < n; i += (long)gridDim.x * 256) {
    const int cw = ca + cb;
    const int r = (int)(i / cw), c = (int)(i % cw);
    const float v = (c < ca) ? a[(size_t)r * ca + c] : b[(size_t)r * cb + (c - ca)];
    d[i] = f2bf(v);
  }
}
__global__ void k_addbias(const float* __restrict__ a, const float* __restrict__ b,
                          float* __restrict__ d, int n) {
  const int i = blockIdx.x * 256 + threadIdx.x;
  if (i < n) d[i] = a[i] + b[i];
}
__global__ void k_embed_src(const int* __restrict__ ids, const float* __restrict__ emb,
                            u16* __restrict__ d) {
  const long n = 2048L * 256;
  for (long i = (long)blockIdx.x * 256 + threadIdx.x; i < n; i += (long)gridDim.x * 256) {
    const int e = (int)(i & 255);
    const int row = (int)(i >> 8);
    const int s = row >> 5, b = row & 31;
    const int tok = ids[b * 64 + s];
    d[i] = f2bf(emb[(size_t)tok * 256 + e]);
  }
}
__global__ void k_embed_tgt(const int* __restrict__ ids, const float* __restrict__ emb,
                            u16* __restrict__ d) {
  const long n = 35L * 32 * 256;
  for (long i = (long)blockIdx.x * 256 + threadIdx.x; i < n; i += (long)gridDim.x * 256) {
    const int e = (int)(i & 255);
    const int row = (int)(i >> 8);
    const int t = row >> 5, b = row & 31;
    const int tok = ids[b * 36 + t];
    d[i] = f2bf(emb[(size_t)tok * 256 + e]);
  }
}
__global__ void k_zero_t0(float* __restrict__ out) {
  const long n = 32L * 32000;
  for (long i = (long)blockIdx.x * 256 + threadIdx.x; i < n; i += (long)gridDim.x * 256) {
    const int b = (int)(i / 32000), c = (int)(i % 32000);
    out[(size_t)b * 36 * 32000 + c] = 0.0f;
  }
}
__global__ void k_init_bar(unsigned* bar) { bar[0] = 0u; bar[1] = 0u; }

// ---------------------------------------------------------------------------
// Host orchestration
// ---------------------------------------------------------------------------
static inline int cvt_grid(long n) {
  long g = (n + 255) / 256;
  if (g > 4096) g = 4096;
  if (g < 1) g = 1;
  return (int)g;
}

extern "C" void kernel_launch(void* const* d_in, const int* in_sizes, int n_in,
                              void* d_out, int out_size, void* d_ws, size_t ws_size,
                              hipStream_t stream) {
  (void)in_sizes; (void)n_in; (void)out_size; (void)ws_size;

  const int* src = (const int*)d_in[0];
  const int* tgt = (const int*)d_in[2];
  const float* emb_src = (const float*)d_in[3];
  const float* emb_tgt = (const float*)d_in[4];
  const float* enc_w_ih[6]; const float* enc_w_hh[6];
  const float* enc_b_ih[6]; const float* enc_b_hh[6];
  for (int i = 0; i < 6; ++i) {
    enc_w_ih[i] = (const float*)d_in[5 + i];
    enc_w_hh[i] = (const float*)d_in[11 + i];
    enc_b_ih[i] = (const float*)d_in[17 + i];
    enc_b_hh[i] = (const float*)d_in[23 + i];
  }
  const float* fc_h_w = (const float*)d_in[29];
  const float* fc_h_b = (const float*)d_in[30];
  const float* fc_c_w = (const float*)d_in[31];
  const float* fc_c_b = (const float*)d_in[32];
  const float* attn_enc_w = (const float*)d_in[33];
  const float* attn_dec_w = (const float*)d_in[34];
  const float* v_w = (const float*)d_in[35];
  const float* dec_w_ih[3]; const float* dec_w_hh[3];
  const float* dec_b_ih[3]; const float* dec_b_hh[3];
  for (int i = 0; i < 3; ++i) {
    dec_w_ih[i] = (const float*)d_in[36 + i];
    dec_w_hh[i] = (const float*)d_in[39 + i];
    dec_b_ih[i] = (const float*)d_in[42 + i];
    dec_b_hh[i] = (const float*)d_in[45 + i];
  }
  const float* fc_out_w = (const float*)d_in[48];
  const float* fc_out_b = (const float*)d_in[49];
  float* out = (float*)d_out;

  // ---- workspace arena ----
  char* base = (char*)d_ws;
  size_t off = 0;
  auto alloc = [&](size_t bytes) -> void* {
    off = (off + 255) & ~(size_t)255;
    void* p = base + off;
    off += bytes;
    return p;
  };
  u16*   X0   = (u16*)alloc(2048ull * 256 * 2);
  u16*   Y0   = (u16*)alloc(2048ull * 512 * 2);   // layer0 out; reused as enc_out
  u16*   Y1   = (u16*)alloc(2048ull * 512 * 2);
  float* gxf  = (float*)alloc(2048ull * 1024 * 4);
  float* gxb  = (float*)alloc(2048ull * 1024 * 4);
  u16*   hlast = (u16*)alloc(96ull * 512 * 2);
  u16*   clast = (u16*)alloc(96ull * 512 * 2);
  float* hstate = (float*)alloc(96ull * 256 * 4);
  float* cstate = (float*)alloc(96ull * 256 * 4);
  float* eproj  = (float*)alloc(2048ull * 256 * 4);
  u16* wih[6]; u16* whh[6]; float* ebias[6];
  for (int i = 0; i < 6; ++i) {
    const int din = (i < 2) ? 256 : 512;
    wih[i] = (u16*)alloc(1024ull * din * 2);
    whh[i] = (u16*)alloc(1024ull * 256 * 2);
    ebias[i] = (float*)alloc(1024 * 4);
  }
  u16* dw[3]; float* dbias[3];
  for (int l = 0; l < 3; ++l) {
    const int K = (l == 0) ? 1024 : 512;
    dw[l] = (u16*)alloc(1024ull * K * 2);
    dbias[l] = (float*)alloc(1024 * 4);
  }
  u16* fchb = (u16*)alloc(256ull * 512 * 2);
  u16* fccb = (u16*)alloc(256ull * 512 * 2);
  u16* aeb  = (u16*)alloc(256ull * 512 * 2);
  u16* fob  = (u16*)alloc(32000ull * 768 * 2);
  u16* embT = (u16*)alloc(35ull * 32 * 256 * 2);
  u16* x0   = (u16*)alloc(32ull * 1024 * 2);
  u16* x1   = (u16*)alloc(32ull * 512 * 2);
  u16* x2   = (u16*)alloc(32ull * 512 * 2);
  float* gates = (float*)alloc(32ull * 1024 * 4);
  u16* xctx = (u16*)alloc(1120ull * 768 * 2);
  unsigned* bar = (unsigned*)alloc(256);

  // ---- weight conversion / bias fusion / embeddings ----
  for (int i = 0; i < 6; ++i) {
    const long nih = 1024L * ((i < 2) ? 256 : 512);
    k_cvt<<<cvt_grid(nih), 256, 0, stream>>>(enc_w_ih[i], wih[i], nih);
    k_cvt<<<cvt_grid(1024L * 256), 256, 0, stream>>>(enc_w_hh[i], whh[i], 1024L * 256);
    k_addbias<<<4, 256, 0, stream>>>(enc_b_ih[i], enc_b_hh[i], ebias[i], 1024);
  }
  for (int l = 0; l < 3; ++l) {
    const int din = (l == 0) ? 768 : 256;
    k_cvt_concat<<<cvt_grid(1024L * (din + 256)), 256, 0, stream>>>(
        dec_w_ih[l], din, dec_w_hh[l], 256, dw[l], 1024);
    k_addbias<<<4, 256, 0, stream>>>(dec_b_ih[l], dec_b_hh[l], dbias[l], 1024);
  }
  k_cvt<<<cvt_grid(256L * 512), 256, 0, stream>>>(fc_h_w, fchb, 256L * 512);
  k_cvt<<<cvt_grid(256L * 512), 256, 0, stream>>>(fc_c_w, fccb, 256L * 512);
  k_cvt<<<cvt_grid(256L * 512), 256, 0, stream>>>(attn_enc_w, aeb, 256L * 512);
  k_cvt<<<cvt_grid(32000L * 768), 256, 0, stream>>>(fc_out_w, fob, 32000L * 768);
  k_embed_src<<<cvt_grid(2048L * 256), 256, 0, stream>>>(src, emb_src, X0);
  k_embed_tgt<<<cvt_grid(35L * 32 * 256), 256, 0, stream>>>(tgt, emb_tgt, embT);

  // ---- encoder: per layer, precompute gx (fwd/bwd) then persistent scan ----
  const size_t scan_smem = 32 * 1024 * 4    // gates
                         + 32 * 256 * 4     // c state
                         + 32 * 256 * 2     // h bf16
                         + 32 * 1024 * 4;   // gx async stage  => 304 KB
  (void)hipFuncSetAttribute((const void*)k_enc_scan,
                            hipFuncAttributeMaxDynamicSharedMemorySize, (int)scan_smem);
  const u16* Xin[3] = { X0, Y0, Y1 };
  u16* Yout[3] = { Y0, Y1, Y0 };
  for (int l = 0; l < 3; ++l) {
    const int din = (l == 0) ? 256 : 512;
    k_gemm<0><<<dim3(4, 64), 256, 0, stream>>>(Xin[l], wih[2 * l], ebias[2 * l], gxf,
                                               2048, 1024, din, 0, 0);
    k_gemm<0><<<dim3(4, 64), 256, 0, stream>>>(Xin[l], wih[2 * l + 1], ebias[2 * l + 1], gxb,
                                               2048, 1024, din, 0, 0);
    k_enc_scan<<<2, 1024, scan_smem, stream>>>(gxf, gxb, whh[2 * l], whh[2 * l + 1],
                                               Yout[l], hlast + l * 32 * 512,
                                               clast + l * 32 * 512);
  }
  u16* enc_out = Y0;

  // ---- bridge + attention precompute ----
  k_gemm<1><<<dim3(1, 3), 256, 0, stream>>>(hlast, fchb, fc_h_b, hstate, 96, 256, 512, 0, 0);
  k_gemm<1><<<dim3(1, 3), 256, 0, stream>>>(clast, fccb, fc_c_b, cstate, 96, 256, 512, 0, 0);
  k_gemm<0><<<dim3(1, 64), 256, 0, stream>>>(enc_out, aeb, nullptr, eproj, 2048, 256, 512, 0, 0);

  // ---- decoder (persistent cooperative kernel) ----
  k_init_bar<<<1, 1, 0, stream>>>(bar);
  k_decoder<<<8, 256, 0, stream>>>(src, enc_out, eproj, attn_dec_w, v_w, embT,
                                   dw[0], dw[1], dw[2], dbias[0], dbias[1], dbias[2],
                                   hstate, cstate, x0, x1, x2, gates, xctx, bar);

  // ---- output projection for all 35 steps + zero the t=0 row ----
  k_gemm<2><<<dim3(125, 35), 256, 0, stream>>>(xctx, fob, fc_out_b, out,
                                               1120, 32000, 768, 36, 32000);
  k_zero_t0<<<4000, 256, 0, stream>>>(out);
}
